// Linear8bit_18021682774289
// MI455X (gfx1250) — matrix operand order repaired and checked by reading, not compiled, not run on hardware
//
#include <hip/hip_runtime.h>
#include <hip/hip_bf16.h>

typedef __attribute__((ext_vector_type(16))) _Float16 v16h;
typedef __attribute__((ext_vector_type(8)))  _Float16 v8h;
typedef __attribute__((ext_vector_type(4)))  _Float16 v4h;
typedef __attribute__((ext_vector_type(8)))  float    v8f;

#define TM 128        // M tile per block
#define TN 128        // N tile per block
#define TK 32         // K step (one wmma_f32_16x16x32_f16 K-slab)
#define LDSP 40       // padded LDS row length in halfs (80B = 20-bank stride)

// out[M,N] = (x[M,K] (f32->f16) @ w_q[N,K] (int8-in-int32 -> f16, exact)) * scale[n] + bias[n]
__global__ __launch_bounds__(256)
void linear8_wmma_f16(const float* __restrict__ x,
                      const int*   __restrict__ wq,
                      const float* __restrict__ scale,
                      const float* __restrict__ bias,
                      float* __restrict__ out,
                      int M, int N, int K)
{
    __shared__ _Float16 xs[2][TM][LDSP];   // A tile: [m][k], f16
    __shared__ _Float16 ws[2][TN][LDSP];   // B tile: [n][k], f16 (weight is [N,K] row-major)

    const int tid   = threadIdx.x;
    const int lane  = tid & 31;
    const int wave  = tid >> 5;            // 0..7
    const int nTile = blockIdx.x * TN;
    const int mTile = blockIdx.y * TM;

    const int mWave = (wave >> 1) * 32;    // 4 wave-rows of 32 M
    const int nWave = (wave & 1) * 64;     // 2 wave-cols of 64 N

    // Global staging: 128 rows x 32 cols per tile; 8 threads/row (8 f32 each via
    // two... actually 4 elems per thread-slot), 32 rows per pass, 4 passes.
    const int gRow = tid >> 3;             // 0..31
    const int gCol = (tid & 7) * 4;        // 0,4,...,28 (elements)

    v8f acc[2][4];
    const v8f vzero = {0.f,0.f,0.f,0.f,0.f,0.f,0.f,0.f};
#pragma unroll
    for (int m = 0; m < 2; ++m)
#pragma unroll
        for (int n = 0; n < 4; ++n)
            acc[m][n] = vzero;

    float4 xr[4];
    int4   wr[4];

    auto gload = [&](int kt) {
        const float* xp = x  + (long long)(mTile + gRow) * K + (long long)kt * TK + gCol;
        const int*   wp = wq + (long long)(nTile + gRow) * K + (long long)kt * TK + gCol;
#pragma unroll
        for (int p = 0; p < 4; ++p) {
            xr[p] = *(const float4*)(xp + (long long)p * 32 * K);
            wr[p] = *(const int4*)  (wp + (long long)p * 32 * K);
        }
    };

    auto stlds = [&](int buf) {
#pragma unroll
        for (int p = 0; p < 4; ++p) {
            v4h hx = { (_Float16)xr[p].x, (_Float16)xr[p].y,
                       (_Float16)xr[p].z, (_Float16)xr[p].w };
            v4h hw = { (_Float16)wr[p].x, (_Float16)wr[p].y,
                       (_Float16)wr[p].z, (_Float16)wr[p].w };
            *(v4h*)&xs[buf][gRow + p * 32][gCol] = hx;
            *(v4h*)&ws[buf][gRow + p * 32][gCol] = hw;
        }
    };

    const int r16 = lane & 15;
    const int hi  = lane >> 4;             // 0 for lanes 0-15, 1 for lanes 16-31

    auto compute = [&](int buf) {
        v16h A[2], B[4];
        // A 16x32 f16 layout: lanes<16 hold K {0..7,16..23}; lanes>=16 hold K {8..15,24..31}
#pragma unroll
        for (int m = 0; m < 2; ++m) {
            const _Float16* base = &xs[buf][mWave + m * 16 + r16][hi * 8];
            union { v16h v; v8h h[2]; } u;
            u.h[0] = *(const v8h*)(base);
            u.h[1] = *(const v8h*)(base + 16);
            A[m] = u.v;
        }
        // B 32x16 f16 layout: lane = N; lanes<16 hold K {0..7,16..23}; lanes>=16 hold K {8..15,24..31}
#pragma unroll
        for (int n = 0; n < 4; ++n) {
            const _Float16* base = &ws[buf][nWave + n * 16 + r16][hi * 8];
            union { v16h v; v8h h[2]; } u;
            u.h[0] = *(const v8h*)(base);
            u.h[1] = *(const v8h*)(base + 16);
            B[n] = u.v;
        }
#pragma unroll
        for (int m = 0; m < 2; ++m)
#pragma unroll
            for (int n = 0; n < 4; ++n)
                acc[m][n] = __builtin_amdgcn_wmma_f32_16x16x32_f16(
                    /*neg_a=*/false, A[m], /*neg_b=*/false, B[n],
                    /*c_mod=*/(short)0, acc[m][n],
                    /*reuse_a=*/false, /*reuse_b=*/false);
    };

    const int nK = K / TK;

    gload(0);
    stlds(0);
    __syncthreads();

    for (int kt = 0; kt < nK; ++kt) {
        const int buf = kt & 1;
        if (kt + 2 < nK) {
            // stream weights: prefetch two K-slabs ahead (global_prefetch_b8)
            __builtin_prefetch(wq + (long long)(nTile + gRow) * K +
                               (long long)(kt + 2) * TK + gCol, 0, 1);
        }
        if (kt + 1 < nK) gload(kt + 1);    // overlap global loads with WMMAs
        compute(buf);
        __syncthreads();                   // everyone done reading lds[buf]
        if (kt + 1 < nK) {
            stlds(buf ^ 1);
            __syncthreads();               // lds[buf^1] ready for next iter
        }
    }

    // Epilogue: D layout per ISA: VGPR r -> M = r (lanes 0-15) / r+8 (lanes 16-31),
    // N = lane & 15. Apply factored-out scale and bias here.
#pragma unroll
    for (int n = 0; n < 4; ++n) {
        const int col = nTile + nWave + n * 16 + r16;
        const float sc = scale[col];
        const float bi = bias[col];
#pragma unroll
        for (int m = 0; m < 2; ++m) {
            const int rowBase = mTile + mWave + m * 16 + (hi << 3);
#pragma unroll
            for (int rr = 0; rr < 8; ++rr) {
                out[(long long)(rowBase + rr) * N + col] = acc[m][n][rr] * sc + bi;
            }
        }
    }
}

extern "C" void kernel_launch(void* const* d_in, const int* in_sizes, int n_in,
                              void* d_out, int out_size, void* d_ws, size_t ws_size,
                              hipStream_t stream) {
    const float* x     = (const float*)d_in[0];   // [B,S,K] f32
    const int*   wq    = (const int*)  d_in[1];   // [N,K] int32 (int8-valued)
    const float* scale = (const float*)d_in[2];   // [N]
    const float* bias  = (const float*)d_in[3];   // [N]
    float*       out   = (float*)d_out;           // [B,S,N] f32

    const int N = in_sizes[2];                    // scale has N elements
    const int K = (int)((long long)in_sizes[1] / N);
    const int M = (int)((long long)in_sizes[0] / K);   // B*S

    dim3 grid(N / TN, M / TM);                    // (128, 8) for the reference shapes
    dim3 block(256);
    linear8_wmma_f16<<<grid, block, 0, stream>>>(x, wq, scale, bias, out, M, N, K);
}
